// WritePath_63058709840237
// MI455X (gfx1250) — matrix operand order repaired and checked by reading, not compiled, not run on hardware
//
#include <hip/hip_runtime.h>

// Problem sizes (compile-time, from the reference).
#define B_ 4
#define T_ 4096
#define H_ 2048
#define D_ 512
#define M_ 8192
#define N_ (B_ * T_)      // 16384 rows
#define CCOLS 1536        // [W_obs | gw1 | pw1]
#define KT_H (H_ / 32)    // 64  K-steps for GEMM-1
#define KT_D (D_ / 32)    // 16  K-steps for matcher
#define NT_W (CCOLS / 16) // 96  col tiles of Wcat
#define MT_  (M_ / 16)    // 512 col tiles of beliefs

typedef __attribute__((ext_vector_type(16))) __bf16       v16bf;
typedef __attribute__((ext_vector_type(8)))  float        v8f;
typedef __attribute__((ext_vector_type(8)))  unsigned int v8u;

union ABFrag { v16bf h; v8u u; };

__device__ __forceinline__ unsigned short f2bf(float f) {
  unsigned int u = __float_as_uint(f);
  u += 0x7FFFu + ((u >> 16) & 1u);   // round-to-nearest-even
  return (unsigned short)(u >> 16);
}
__device__ __forceinline__ unsigned int pack2bf(float a, float b) {
  return (unsigned int)f2bf(a) | ((unsigned int)f2bf(b) << 16);
}
__device__ __forceinline__ v8f zero8() {
  v8f z;
  #pragma unroll
  for (int i = 0; i < 8; ++i) z[i] = 0.0f;
  return z;
}
// A-operand K offset for (v, half) per ISA 16-bit A 16x32 layout.
__device__ __forceinline__ int a_koff(int v, int hf) {
  return (v < 4) ? (2 * v + 8 * hf) : (16 + 2 * (v - 4) + 8 * hf);
}
// LDS byte offset of a shared-memory pointer (flat addr low 32 bits).
__device__ __forceinline__ unsigned lds_off(const void* p) {
  return (unsigned)(size_t)p;
}
// CDNA5 async global->LDS copy (ASYNCcnt tracked, no VGPR data).
__device__ __forceinline__ void async_copy_b128(unsigned dst_lds, const void* src) {
  asm volatile("global_load_async_to_lds_b128 %0, %1, off"
               :: "v"(dst_lds), "v"((unsigned long long)src) : "memory");
}
__device__ __forceinline__ void wait_async0() {
  asm volatile("s_wait_asynccnt 0" ::: "memory");
}

// ---------------------------------------------------------------- prep ----
__global__ void k_zero(float* p, int n) {
  int i = blockIdx.x * blockDim.x + threadIdx.x;
  if (i < n) p[i] = 0.0f;
}

// hidden f32 [N,H] -> A-fragment-major bf16 dwords:
// haswz[(((rowTile*KT_H)+kt)*32 + lane)*8 + v]
__global__ void k_pack_hidden(const float* __restrict__ src,
                              unsigned int* __restrict__ dst) {
  int idx = blockIdx.x * blockDim.x + threadIdx.x;   // N_*H_/2 dwords
  int v = idx & 7;
  int lane = (idx >> 3) & 31;
  int kt = (idx >> 8) & (KT_H - 1);
  int rowTile = idx >> 14;
  int r16 = lane & 15, hf = lane >> 4;
  int row = rowTile * 16 + r16;
  int k0 = kt * 32 + a_koff(v, hf);
  const float* p = src + (size_t)row * H_ + k0;
  dst[idx] = pack2bf(p[0], p[1]);
}

// [W_obs|gw1|pw1] -> B-fragment-major dwords:
// wswz[((kt*NT_W + nt)*32 + lane)*8 + v], B layout k0 = a_koff(v, hf).
__global__ void k_pack_w(const float* __restrict__ W_obs,
                         const float* __restrict__ gw1,
                         const float* __restrict__ pw1,
                         unsigned int* __restrict__ wswz) {
  int idx = blockIdx.x * blockDim.x + threadIdx.x;   // KT_H*NT_W*256 dwords
  if (idx >= KT_H * NT_W * 256) return;
  int kt = idx / (NT_W * 256);
  int rem = idx % (NT_W * 256);
  int nt = rem >> 8;
  int lane = (rem >> 3) & 31;
  int v = rem & 7;
  int hf = lane >> 4;
  int c = nt * 16 + (lane & 15);
  int k0 = kt * 32 + a_koff(v, hf);
  const float* Wsel;
  int cc = c;
  if (c < 512)       { Wsel = W_obs; }
  else if (c < 1024) { Wsel = gw1; cc = c - 512; }
  else               { Wsel = pw1; cc = c - 1024; }
  wswz[idx] = pack2bf(Wsel[(size_t)k0 * 512 + cc], Wsel[(size_t)(k0 + 1) * 512 + cc]);
}

// Belief row inverse norms.
__global__ void k_bnorm(const float* __restrict__ beliefs, float* __restrict__ bninv) {
  int m = blockIdx.x;
  int tid = threadIdx.x;
  __shared__ float red[256];
  float v0 = beliefs[(size_t)m * 512 + tid];
  float v1 = beliefs[(size_t)m * 512 + 256 + tid];
  red[tid] = v0 * v0 + v1 * v1;
  __syncthreads();
  for (int s = 128; s > 0; s >>= 1) {
    if (tid < s) red[tid] += red[tid + s];
    __syncthreads();
  }
  if (tid == 0) bninv[m] = 1.0f / fmaxf(sqrtf(red[0]), 1e-8f);
}

// Normalized beliefs^T -> B-fragment-major, per-mt contiguous 16KB slabs:
// bswz[((mt*KT_D + ks)*32 + lane)*8 + v]
__global__ void k_pack_b(const float* __restrict__ beliefs,
                         const float* __restrict__ bninv,
                         unsigned int* __restrict__ bswz) {
  int idx = blockIdx.x * blockDim.x + threadIdx.x;   // MT_*KT_D*256 dwords
  int v = idx & 7;
  int lane = (idx >> 3) & 31;
  int ks = (idx >> 8) & (KT_D - 1);
  int mt = idx >> 12;
  int hf = lane >> 4;
  int n = mt * 16 + (lane & 15);
  int k0 = ks * 32 + a_koff(v, hf);
  float inv = bninv[n];
  bswz[idx] = pack2bf(beliefs[(size_t)n * 512 + k0] * inv,
                      beliefs[(size_t)n * 512 + k0 + 1] * inv);
}

// ------------------------------------------------- GEMM 1: hidden @ Wcat ----
// Block = 8 waves sharing one 64-col tile; B slice double-buffered in LDS via
// async global->LDS loads.  wave w handles rows (rowGroup*8+w)*16 .. +15.
__global__ void __launch_bounds__(256)
k_gemm_obs(const unsigned int* __restrict__ haswz,
           const unsigned int* __restrict__ wswz,
           const float* __restrict__ gb1, const float* __restrict__ gw2,
           const float* __restrict__ pb1, const float* __restrict__ pw2,
           float* __restrict__ obs, float* gacc, float* pacc) {
  __shared__ __align__(16) unsigned int bsh[2][1024];   // 2 x 4KB slabs
  const int wid = threadIdx.x >> 5;
  const int lane = threadIdx.x & 31;
  const int colTile = blockIdx.x % 24;               // 64-col tile
  const int rowGroup = blockIdx.x / 24;              // 0..127
  const int rowTile = rowGroup * 8 + wid;
  const int r16 = lane & 15, hf = lane >> 4;

  const v8u* abase = ((const v8u*)haswz) + ((size_t)rowTile * KT_H * 32 + lane);
  const uint4* wbase = ((const uint4*)wswz) + (size_t)colTile * 256 + threadIdx.x;
  // per-kt stride in uint4 units: NT_W*64
  v8f acc0 = zero8(), acc1 = zero8(), acc2 = zero8(), acc3 = zero8();

  // prime slab 0
  async_copy_b128(lds_off(&bsh[0][threadIdx.x * 4]), wbase);
  wait_async0();
  __syncthreads();

  for (int kt = 0; kt < KT_H; ++kt) {
    const int cur = kt & 1;
    if (kt + 1 < KT_H)
      async_copy_b128(lds_off(&bsh[cur ^ 1][threadIdx.x * 4]),
                      wbase + (size_t)(kt + 1) * (NT_W * 64));

    ABFrag afr;
    afr.u = abase[(size_t)kt * 32];                  // 2x global_load_b128
    const unsigned int* bs = &bsh[cur][lane * 8];
    ABFrag b0, b1, b2, b3;
    b0.u = *(const v8u*)(bs);
    b1.u = *(const v8u*)(bs + 256);
    b2.u = *(const v8u*)(bs + 512);
    b3.u = *(const v8u*)(bs + 768);
    acc0 = __builtin_amdgcn_wmma_f32_16x16x32_bf16(false, afr.h, false, b0.h,
                                                   (short)0, acc0, false, false);
    acc1 = __builtin_amdgcn_wmma_f32_16x16x32_bf16(false, afr.h, false, b1.h,
                                                   (short)0, acc1, false, false);
    acc2 = __builtin_amdgcn_wmma_f32_16x16x32_bf16(false, afr.h, false, b2.h,
                                                   (short)0, acc2, false, false);
    acc3 = __builtin_amdgcn_wmma_f32_16x16x32_bf16(false, afr.h, false, b3.h,
                                                   (short)0, acc3, false, false);

    wait_async0();        // next slab landed
    __syncthreads();      // all waves done reading cur, next slab published
  }

  #pragma unroll
  for (int ct = 0; ct < 4; ++ct) {
    #pragma unroll
    for (int r = 0; r < 8; ++r) {
      int row = rowTile * 16 + r + 8 * hf;
      int col = colTile * 64 + ct * 16 + r16;
      float v = (ct == 0) ? acc0[r] : (ct == 1) ? acc1[r] : (ct == 2) ? acc2[r] : acc3[r];
      if (col < 512) {
        obs[(size_t)row * 512 + col] = v;
      } else if (col < 1024) {
        float x = fmaxf(v + gb1[col - 512], 0.0f);
        atomicAdd(&gacc[row], x * gw2[col - 512]);
      } else {
        float x = fmaxf(v + pb1[col - 1024], 0.0f);
        atomicAdd(&pacc[row], x * pw2[col - 1024]);
      }
    }
  }
}

// ----------------------------------------------------- row-wise epilogue ----
// One block per row; also writes obs-angle A fragments pre-swizzled:
// aswz[((rowTile*KT_D + kt)*32 + lane)*8 + v]
__global__ void k_rowwise(const float* __restrict__ obs,
                          const float* __restrict__ gacc,
                          const float* __restrict__ pacc,
                          const float* __restrict__ gb2,
                          const float* __restrict__ pb2,
                          float* __restrict__ out_beliefs,
                          unsigned int* __restrict__ aswz,
                          float* __restrict__ meaningful) {
  int row = blockIdx.x;
  int tid = threadIdx.x;                  // 256 threads, 1 dword-pair each
  __shared__ float red[256];
  const float2 vv = ((const float2*)obs)[(size_t)row * 256 + tid];
  red[tid] = vv.x * vv.x + vv.y * vv.y;
  __syncthreads();
  for (int s = 128; s > 0; s >>= 1) {
    if (tid < s) red[tid] += red[tid + s];
    __syncthreads();
  }
  float nrm = sqrtf(red[0]);
  float inv = 1.0f / fmaxf(nrm, 1e-8f);
  float g = gacc[row] + gb2[0];
  float p = pacc[row] + pb2[0];
  float gate = 1.0f / (1.0f + expf(-g));
  float prec = (p > 20.0f) ? p : log1pf(expf(p));
  float gp = gate * prec;
  float a0 = vv.x * inv, a1 = vv.y * inv;
  float2 ob; ob.x = a0 * gp; ob.y = a1 * gp;
  ((float2*)out_beliefs)[(size_t)row * 256 + tid] = ob;

  // scatter bf16 pair into A-fragment-major layout
  int k0 = 2 * tid;
  int kt = k0 >> 5;
  int kk = k0 & 31;
  int hf, v;
  if (kk < 8)       { hf = 0; v = kk >> 1; }
  else if (kk < 16) { hf = 1; v = (kk - 8) >> 1; }
  else if (kk < 24) { hf = 0; v = 4 + ((kk - 16) >> 1); }
  else              { hf = 1; v = 4 + ((kk - 24) >> 1); }
  int lane = (row & 15) + 16 * hf;
  int rowTile = row >> 4;
  aswz[(((size_t)rowTile * KT_D + kt) * 32 + lane) * 8 + v] = pack2bf(a0, a1);

  if (tid == 0) {
    float radius = gp * (nrm * inv);            // == ||obs_beliefs row||
    meaningful[row] = (radius > 0.05f) ? 1.0f : 0.0f;
  }
}

// --------------------------------- GEMM 2: cosine sims + fused argmax ----
// Block = 8 waves x 16 rows each; 16KB belief slab per column tile is
// broadcast through double-buffered LDS via async global->LDS loads.
// A fragments (K=512) live in VGPRs.
__global__ void __launch_bounds__(256)
k_sims_argmax(const unsigned int* __restrict__ aswz,
              const unsigned int* __restrict__ bswz,
              const float* __restrict__ meaningful,
              const int* __restrict__ act_idx,
              float* __restrict__ slots,
              float* __restrict__ sims_out,
              float* __restrict__ matched_out) {
  __shared__ __align__(16) unsigned int bsh[2][4096];   // 2 x 16KB slabs
  const int wid = threadIdx.x >> 5;
  const int lane = threadIdx.x & 31;
  const int rowTile = blockIdx.x * 8 + wid;          // 0..1023
  const int rowbase = rowTile * 16;
  const int r16 = lane & 15, hf = lane >> 4;

  ABFrag afr[KT_D];
  {
    const v8u* ab = ((const v8u*)aswz) + ((size_t)rowTile * KT_D * 32 + lane);
    #pragma unroll
    for (int ks = 0; ks < KT_D; ++ks) afr[ks].u = ab[ks * 32];
  }

  float bestv[8];
  int besti[8];
  #pragma unroll
  for (int r = 0; r < 8; ++r) { bestv[r] = -2.0f; besti[r] = 0; }

  const uint4* bb = ((const uint4*)bswz) + threadIdx.x * 4;
  // prime slab 0 (4 x b128 per thread)
  #pragma unroll
  for (int j = 0; j < 4; ++j)
    async_copy_b128(lds_off(&bsh[0][threadIdx.x * 16 + j * 4]), bb + j);
  wait_async0();
  __syncthreads();

  for (int mt = 0; mt < MT_; ++mt) {
    const int cur = mt & 1;
    if (mt + 1 < MT_) {
      const uint4* src = bb + (size_t)(mt + 1) * 1024;
      #pragma unroll
      for (int j = 0; j < 4; ++j)
        async_copy_b128(lds_off(&bsh[cur ^ 1][threadIdx.x * 16 + j * 4]), src + j);
    }

    v8f acc = zero8();
    const unsigned int* bs = &bsh[cur][lane * 8];
    #pragma unroll
    for (int ks = 0; ks < KT_D; ++ks) {
      ABFrag bfr;
      bfr.u = *(const v8u*)(bs + ks * 256);          // 2x ds_load_b128
      acc = __builtin_amdgcn_wmma_f32_16x16x32_bf16(
          false, afr[ks].h, false, bfr.h, (short)0, acc, false, false);
    }
    int n = mt * 16 + r16;
    #pragma unroll
    for (int r = 0; r < 8; ++r) {
      float v = acc[r];
      if (v > bestv[r]) { bestv[r] = v; besti[r] = n; }
    }

    wait_async0();        // next slab landed
    __syncthreads();      // all waves done reading cur
  }

  // Reduce (max, argidx) across the 16 lanes of each wave32 half.
  #pragma unroll
  for (int r = 0; r < 8; ++r) {
    float v = bestv[r];
    int i = besti[r];
    #pragma unroll
    for (int m = 8; m >= 1; m >>= 1) {
      float ov = __shfl_xor(v, m, 32);
      int oi = __shfl_xor(i, m, 32);
      if (ov > v || (ov == v && oi < i)) { v = ov; i = oi; }
    }
    if (r16 == 0) {
      int row = rowbase + r + 8 * hf;
      bool mn = meaningful[row] > 0.5f;
      bool mat = mn && (v > 0.7f);
      slots[row] = mat ? (float)act_idx[i] : -1.0f;
      sims_out[row] = mat ? v : 0.0f;
      matched_out[row] = mat ? 1.0f : 0.0f;
    }
  }
}

// -------------------------------------------------------------- launcher ----
extern "C" void kernel_launch(void* const* d_in, const int* in_sizes, int n_in,
                              void* d_out, int out_size, void* d_ws, size_t ws_size,
                              hipStream_t stream) {
  const float* hidden  = (const float*)d_in[0];
  const float* beliefs = (const float*)d_in[1];
  const int*   act_idx = (const int*)d_in[2];
  const float* W_obs   = (const float*)d_in[3];
  const float* gw1     = (const float*)d_in[4];
  const float* gb1     = (const float*)d_in[5];
  const float* gw2     = (const float*)d_in[6];
  const float* gb2     = (const float*)d_in[7];
  const float* pw1     = (const float*)d_in[8];
  const float* pb1     = (const float*)d_in[9];
  const float* pw2     = (const float*)d_in[10];
  const float* pb2     = (const float*)d_in[11];

  char* ws = (char*)d_ws;
  size_t off = 0;
  unsigned int* haswz = (unsigned int*)(ws + off); off += (size_t)N_ * H_ / 2 * 4;
  unsigned int* wswz  = (unsigned int*)(ws + off); off += (size_t)KT_H * NT_W * 256 * 4;
  unsigned int* bswz  = (unsigned int*)(ws + off); off += (size_t)MT_ * KT_D * 256 * 4;
  float* obs          = (float*)(ws + off);        off += (size_t)N_ * D_ * 4;
  float* gacc         = (float*)(ws + off);        off += (size_t)N_ * 4;
  float* pacc         = (float*)(ws + off);        off += (size_t)N_ * 4;
  unsigned int* aswz  = (unsigned int*)(ws + off); off += (size_t)N_ * D_ / 2 * 4;
  float* bninv        = (float*)(ws + off);        off += (size_t)M_ * 4;
  float* meaningful   = (float*)(ws + off);        off += (size_t)N_ * 4;
  if (ws_size < off) return;   // workspace too small (constant per run)

  float* outp        = (float*)d_out;
  float* out_obs     = outp;                          // [N, D]
  float* out_slots   = outp + (size_t)N_ * D_;        // [N]
  float* out_sims    = out_slots + N_;                // [N]
  float* out_matched = out_sims + N_;                 // [N]

  // 1) zero head accumulators (gacc and pacc are adjacent)
  k_zero<<<(2 * N_ + 255) / 256, 256, 0, stream>>>(gacc, 2 * N_);
  // 2) hidden -> A-fragment-major bf16
  k_pack_hidden<<<(N_ * H_ / 2) / 256, 256, 0, stream>>>(hidden, haswz);
  // 3) concatenated weights -> B-fragment-major bf16
  k_pack_w<<<(KT_H * NT_W * 256 + 255) / 256, 256, 0, stream>>>(W_obs, gw1, pw1, wswz);
  // 4) belief norms, then normalized beliefs^T -> B-fragment-major slabs
  k_bnorm<<<M_, 256, 0, stream>>>(beliefs, bninv);
  k_pack_b<<<(MT_ * KT_D * 256) / 256, 256, 0, stream>>>(beliefs, bninv, bswz);
  // 5) big fused GEMM (obs + both head first layers)
  k_gemm_obs<<<128 * 24, 256, 0, stream>>>(haswz, wswz, gb1, gw2, pb1, pw2,
                                           obs, gacc, pacc);
  // 6) per-row normalize + gate/precision epilogue (+ A-fragment scatter)
  k_rowwise<<<N_, 256, 0, stream>>>(obs, gacc, pacc, gb2, pb2,
                                    out_obs, aswz, meaningful);
  // 7) matching GEMM + fused top-1 argmax
  k_sims_argmax<<<(N_ / 16) / 8, 256, 0, stream>>>(aswz, bswz, meaningful, act_idx,
                                                   out_slots, out_sims, out_matched);
}
